// DecoupledBottleneckAttention_10617159155957
// MI455X (gfx1250) — matrix operand order repaired and checked by reading, not compiled, not run on hardware
//
#include <hip/hip_runtime.h>
#include <hip/hip_bf16.h>
#include <stdint.h>

#define H_ 16
#define T_ 4096
#define DM_ 1024
#define QKVN_ 1280
#define AD_ 512

typedef __bf16 bf16_t;
typedef bf16_t v16bf __attribute__((ext_vector_type(16)));
typedef bf16_t v8bf  __attribute__((ext_vector_type(8)));
typedef float  v8f   __attribute__((ext_vector_type(8)));
typedef int    v4i   __attribute__((ext_vector_type(4)));

static __device__ __forceinline__ bf16_t f2bf(float f) {
  unsigned u = __float_as_uint(f);
  unsigned r = (u + 0x7FFFu + ((u >> 16) & 1u)) >> 16;
  unsigned short hs = (unsigned short)r;
  return __builtin_bit_cast(bf16_t, hs);
}

#define WMMA_BF16(A_, B_, C_) \
  __builtin_amdgcn_wmma_f32_16x16x32_bf16(false, (A_), false, (B_), (short)0, (C_), false, false)

// ---------------- CDNA5 async global->LDS path (guarded; falls back to reg staging)
#if defined(__has_builtin)
#if __has_builtin(__builtin_amdgcn_global_load_async_to_lds_b128)
#define HAS_ASYNC_LDS 1
#endif
#endif
#ifndef HAS_ASYNC_LDS
#define HAS_ASYNC_LDS 0
#endif

#if HAS_ASYNC_LDS
typedef __attribute__((address_space(1))) v4i as1_v4i;   // global int4*
typedef __attribute__((address_space(3))) v4i as3_v4i;   // LDS int4*
static __device__ __forceinline__ void async_g2l_b128(const void* g, void* l) {
  // per-lane 16B transfer, global -> LDS, tracked by ASYNCcnt
  __builtin_amdgcn_global_load_async_to_lds_b128((as1_v4i*)g, (as3_v4i*)l, 0, 0);
}
static __device__ __forceinline__ void wait_async0() {
#if __has_builtin(__builtin_amdgcn_s_wait_asynccnt)
  __builtin_amdgcn_s_wait_asynccnt(0);
#else
  asm volatile("s_wait_asynccnt 0x0" ::: "memory");
#endif
}
#endif

// ---------------------------------------------------------------- convert f32 -> bf16
__global__ void k_cvt_f32_to_bf16(const float* __restrict__ src,
                                  bf16_t* __restrict__ dst, int n) {
  int i = blockIdx.x * blockDim.x + threadIdx.x;
  if (i < n) dst[i] = f2bf(src[i]);
}

// ---------------------------------------------------------------- bf16 GEMM  C[M][N] = A[M][K] * B[N][K]^T
#define TM 128
#define TN 128
#define TK 32

__global__ __launch_bounds__(256)
void k_gemm_bf16(const bf16_t* __restrict__ A, const bf16_t* __restrict__ B,
                 float* __restrict__ C, int M, int N, int K)
{
  __shared__ bf16_t As[2][TM][TK];   // [m][k]
  __shared__ bf16_t Bs[2][TN][TK];   // [n][k]

  const int tid  = threadIdx.x;
  const int lane = tid & 31;
  const int warp = tid >> 5;
  const int ln   = lane & 15;
  const int half = lane >> 4;

  const int ntn = N / TN;
  const int m0  = (blockIdx.x / ntn) * TM;
  const int n0  = (blockIdx.x % ntn) * TN;

  const int wm = (warp & 3) * 32;   // wave: 32 rows x 64 cols of the tile
  const int wn = (warp >> 2) * 64;

  const int lrow = tid >> 1;        // 256 threads stage 128x32 bf16 (16 elems each)
  const int lkh  = (tid & 1) * 16;

  const bf16_t* ag = A + (size_t)(m0 + lrow) * K + lkh;
  const bf16_t* bg = B + (size_t)(n0 + lrow) * K + lkh;

  v8f acc[2][4];
  #pragma unroll
  for (int i = 0; i < 2; ++i)
    #pragma unroll
    for (int j = 0; j < 4; ++j)
      #pragma unroll
      for (int r = 0; r < 8; ++r) acc[i][j][r] = 0.0f;

  // ---- stage tile 0
#if HAS_ASYNC_LDS
  async_g2l_b128(ag,     &As[0][lrow][lkh]);
  async_g2l_b128(ag + 8, &As[0][lrow][lkh + 8]);
  async_g2l_b128(bg,     &Bs[0][lrow][lkh]);
  async_g2l_b128(bg + 8, &Bs[0][lrow][lkh + 8]);
  wait_async0();
  __syncthreads();
#else
  {
    v16bf aReg = *(const v16bf*)ag;
    v16bf bReg = *(const v16bf*)bg;
    *(v16bf*)&As[0][lrow][lkh] = aReg;
    *(v16bf*)&Bs[0][lrow][lkh] = bReg;
  }
  __syncthreads();
#endif

  const int ksteps = K / TK;
  int buf = 0;
  for (int ks = 0; ks < ksteps; ++ks) {
#if HAS_ASYNC_LDS
    if (ks + 1 < ksteps) {       // issue async copies for next tile; overlap with WMMA
      const bf16_t* an = ag + (size_t)(ks + 1) * TK;
      const bf16_t* bn = bg + (size_t)(ks + 1) * TK;
      async_g2l_b128(an,     &As[buf ^ 1][lrow][lkh]);
      async_g2l_b128(an + 8, &As[buf ^ 1][lrow][lkh + 8]);
      async_g2l_b128(bn,     &Bs[buf ^ 1][lrow][lkh]);
      async_g2l_b128(bn + 8, &Bs[buf ^ 1][lrow][lkh + 8]);
    }
#else
    v16bf aN, bN;
    if (ks + 1 < ksteps) {
      aN = *(const v16bf*)(ag + (size_t)(ks + 1) * TK);
      bN = *(const v16bf*)(bg + (size_t)(ks + 1) * TK);
    }
#endif
    // A fragments (16x32 bf16): lane ln = row, K = half*16 + 0..15 contiguous
    v16bf af[2];
    #pragma unroll
    for (int rt = 0; rt < 2; ++rt) {
      const bf16_t* ap = &As[buf][wm + rt * 16 + ln][half * 16];
      v8bf lo = *(const v8bf*)ap;
      v8bf hi = *(const v8bf*)(ap + 8);
      #pragma unroll
      for (int i = 0; i < 8; ++i) { af[rt][i] = lo[i]; af[rt][8 + i] = hi[i]; }
    }
    // B fragments (32x16 bf16): lane ln = col, K = half*16 + 0..15 contiguous
    #pragma unroll
    for (int ct = 0; ct < 4; ++ct) {
      v16bf bfv = *(const v16bf*)&Bs[buf][wn + ct * 16 + ln][half * 16];
      #pragma unroll
      for (int rt = 0; rt < 2; ++rt)
        acc[rt][ct] = WMMA_BF16(af[rt], bfv, acc[rt][ct]);
    }
#if HAS_ASYNC_LDS
    wait_async0();               // this wave's async tile writes have landed
    __syncthreads();             // all waves' writes visible before next read
#else
    if (ks + 1 < ksteps) {
      *(v16bf*)&As[buf ^ 1][lrow][lkh] = aN;
      *(v16bf*)&Bs[buf ^ 1][lrow][lkh] = bN;
    }
    __syncthreads();
#endif
    buf ^= 1;
  }

  // C/D layout: lane element r -> row r + 8*half, col ln
  #pragma unroll
  for (int rt = 0; rt < 2; ++rt)
    #pragma unroll
    for (int ct = 0; ct < 4; ++ct) {
      float* cp = C + (size_t)(m0 + wm + rt * 16 + 8 * half) * N + (n0 + wn + ct * 16 + ln);
      #pragma unroll
      for (int r = 0; r < 8; ++r) cp[(size_t)r * N] = acc[rt][ct][r];
    }
}

// ---------------------------------------------------------------- RoPE + pack (pad QK head dim 24->32)
__global__ __launch_bounds__(256)
void k_pack_rope(const float* __restrict__ QKV,          // [T][1280]
                 const float* __restrict__ logit_scale,  // [16]
                 const int* __restrict__ pos_offset_p,
                 bf16_t* __restrict__ Qb,                 // [H][T][32], scaled
                 bf16_t* __restrict__ Kb,                 // [H][T][32]
                 bf16_t* __restrict__ Vt)                 // [H][32][T]
{
  int idx = blockIdx.x * blockDim.x + threadIdx.x;
  int h = idx & (H_ - 1);
  int t = idx >> 4;
  if (t >= T_) return;

  const float* row = QKV + (size_t)t * QKVN_;
  float pos = (float)(pos_offset_p[0] + t);
  float qscale = __expf(logit_scale[h]) * 0.20412414523193154f;  // 1/sqrt(24)

  bf16_t* qd = Qb + ((size_t)h * T_ + t) * 32;
  bf16_t* kd = Kb + ((size_t)h * T_ + t) * 32;

  #pragma unroll
  for (int i = 0; i < 8; ++i) {                // semantic dims
    qd[i] = f2bf(row[h * 8 + i] * qscale);
    kd[i] = f2bf(row[128 + h * 8 + i]);
  }
  const float* qg = row + 256 + h * 16;        // geometric dims + RoPE (rot=16)
  const float* kg = row + 512 + h * 16;
  #pragma unroll
  for (int i = 0; i < 8; ++i) {
    float inv = __expf(-(2.0f * (float)i / 16.0f) * 9.210340371976184f); // base^(-2i/16)
    float fr = pos * inv;
    float sn, cs;
    __sincosf(fr, &sn, &cs);
    float q1 = qg[i], q2 = qg[i + 8];
    qd[8 + i]  = f2bf((q1 * cs - q2 * sn) * qscale);
    qd[16 + i] = f2bf((q1 * sn + q2 * cs) * qscale);
    float k1 = kg[i], k2 = kg[i + 8];
    kd[8 + i]  = f2bf(k1 * cs - k2 * sn);
    kd[16 + i] = f2bf(k1 * sn + k2 * cs);
  }
  #pragma unroll
  for (int i = 0; i < 8; ++i) { qd[24 + i] = f2bf(0.0f); kd[24 + i] = f2bf(0.0f); }

  const float* vv = row + 768 + h * 32;
  #pragma unroll
  for (int d = 0; d < 32; ++d)
    Vt[((size_t)h * 32 + d) * T_ + t] = f2bf(vv[d]);
}

// ---------------------------------------------------------------- causal flash attention, 1 wave = 16 queries of one head
__global__ __launch_bounds__(256)
void k_attn(const bf16_t* __restrict__ Qb, const bf16_t* __restrict__ Kb,
            const bf16_t* __restrict__ Vt, bf16_t* __restrict__ Ob /* [T][512] */)
{
  __shared__ bf16_t Pld[8][16][32];   // per-wave P transpose buffer

  const int lane = threadIdx.x & 31;
  const int warp = threadIdx.x >> 5;
  const int g  = blockIdx.x * 8 + warp;
  const int h  = g >> 8;            // T/16 = 256 query tiles per head
  const int qt = g & 255;
  const int q0 = qt * 16;
  const int ln   = lane & 15;
  const int half = lane >> 4;

  // Q A-fragment: row = ln (query q0+ln), K = half*16 + 0..15 contiguous
  v16bf qa;
  {
    const bf16_t* qp = Qb + ((size_t)h * T_ + q0 + ln) * 32 + half * 16;
    v8bf lo = *(const v8bf*)qp;
    v8bf hi = *(const v8bf*)(qp + 8);
    #pragma unroll
    for (int i = 0; i < 8; ++i) { qa[i] = lo[i]; qa[8 + i] = hi[i]; }
  }

  v8f o0, o1;
  float mrow[8], lrow[8];
  #pragma unroll
  for (int r = 0; r < 8; ++r) { o0[r] = 0.0f; o1[r] = 0.0f; mrow[r] = -3.0e38f; lrow[r] = 0.0f; }

  const bf16_t* kbase_p = Kb + (size_t)h * T_ * 32;
  const bf16_t* vbase_p = Vt + (size_t)h * 32 * T_;

  const int nkb = (q0 + 47) >> 5;   // key blocks of 32 covering keys 0..q0+15
  for (int kb = 0; kb < nkb; ++kb) {
    const int kbase = kb * 32;

    // K B-fragments: lane col ln = key index, K(dim) = half*16 + 0..15 contiguous
    v16bf bk0 = *(const v16bf*)(kbase_p + (size_t)(kbase + ln) * 32 + half * 16);
    v16bf bk1 = *(const v16bf*)(kbase_p + (size_t)(kbase + 16 + ln) * 32 + half * 16);

    v8f s0, s1;
    #pragma unroll
    for (int r = 0; r < 8; ++r) { s0[r] = 0.0f; s1[r] = 0.0f; }
    s0 = WMMA_BF16(qa, bk0, s0);
    s1 = WMMA_BF16(qa, bk1, s1);

    // causal mask + row max (rows live in 16-lane halves; xor 1/2/4/8 stays in-half)
    float pmax[8];
    #pragma unroll
    for (int r = 0; r < 8; ++r) {
      int q = q0 + r + 8 * half;
      if (kbase + ln > q)      s0[r] = -3.0e38f;
      if (kbase + 16 + ln > q) s1[r] = -3.0e38f;
      float mx = fmaxf(s0[r], s1[r]);
      #pragma unroll
      for (int off = 1; off < 16; off <<= 1) mx = fmaxf(mx, __shfl_xor(mx, off, 32));
      pmax[r] = mx;
    }
    // online softmax update
    #pragma unroll
    for (int r = 0; r < 8; ++r) {
      float mnew  = fmaxf(mrow[r], pmax[r]);
      float alpha = __expf(mrow[r] - mnew);
      float p0 = __expf(s0[r] - mnew);
      float p1 = __expf(s1[r] - mnew);
      s0[r] = p0; s1[r] = p1;
      float sum = p0 + p1;
      #pragma unroll
      for (int off = 1; off < 16; off <<= 1) sum += __shfl_xor(sum, off, 32);
      lrow[r] = lrow[r] * alpha + sum;
      mrow[r] = mnew;
      o0[r] *= alpha; o1[r] *= alpha;
    }
    // transpose P (C-layout -> A-layout) through per-wave LDS (DS ops are in-order per wave)
    #pragma unroll
    for (int r = 0; r < 8; ++r) {
      Pld[warp][r + 8 * half][ln]      = f2bf(s0[r]);
      Pld[warp][r + 8 * half][16 + ln] = f2bf(s1[r]);
    }
    v16bf pa;
    {
      const bf16_t* pp = &Pld[warp][ln][half * 16];
      v8bf lo = *(const v8bf*)pp;
      v8bf hi = *(const v8bf*)(pp + 8);
      #pragma unroll
      for (int i = 0; i < 8; ++i) { pa[i] = lo[i]; pa[8 + i] = hi[i]; }
    }
    // V B-fragments from Vt[H][32][T]: lane col ln = vdim, K(key) = kbase + half*16 + 0..15
    v16bf bv0 = *(const v16bf*)(vbase_p + (size_t)ln * T_ + kbase + half * 16);
    v16bf bv1 = *(const v16bf*)(vbase_p + (size_t)(16 + ln) * T_ + kbase + half * 16);
    o0 = WMMA_BF16(pa, bv0, o0);
    o1 = WMMA_BF16(pa, bv1, o1);
  }

  // normalize and store O as bf16 [T][H*32] for the output projection
  #pragma unroll
  for (int r = 0; r < 8; ++r) {
    int q = q0 + r + 8 * half;
    float inv = 1.0f / lrow[r];
    Ob[(size_t)q * AD_ + h * 32 + ln]      = f2bf(o0[r] * inv);
    Ob[(size_t)q * AD_ + h * 32 + 16 + ln] = f2bf(o1[r] * inv);
  }
}

// ---------------------------------------------------------------- host
extern "C" void kernel_launch(void* const* d_in, const int* in_sizes, int n_in,
                              void* d_out, int out_size, void* d_ws, size_t ws_size,
                              hipStream_t stream)
{
  (void)in_sizes; (void)n_in; (void)out_size; (void)ws_size;
  const float* x    = (const float*)d_in[0];
  const float* qsw  = (const float*)d_in[1];
  const float* ksw  = (const float*)d_in[2];
  const float* qgw  = (const float*)d_in[3];
  const float* kgw  = (const float*)d_in[4];
  const float* vw   = (const float*)d_in[5];
  const float* ow   = (const float*)d_in[6];
  const float* lsc  = (const float*)d_in[7];
  const int*   poff = (const int*)d_in[9];
  float* out = (float*)d_out;

  uint8_t* ws = (uint8_t*)d_ws;
  size_t off = 0;
  auto alloc = [&](size_t bytes) {
    uint8_t* p = ws + off;
    off = (off + bytes + 255) & ~(size_t)255;
    return p;
  };
  bf16_t* xb  = (bf16_t*)alloc((size_t)T_ * DM_ * 2);
  bf16_t* Wb  = (bf16_t*)alloc((size_t)QKVN_ * DM_ * 2);
  bf16_t* OWb = (bf16_t*)alloc((size_t)DM_ * AD_ * 2);
  float*  QKV = (float*)alloc((size_t)T_ * QKVN_ * 4);
  bf16_t* Qb  = (bf16_t*)alloc((size_t)H_ * T_ * 32 * 2);
  bf16_t* Kb  = (bf16_t*)alloc((size_t)H_ * T_ * 32 * 2);
  bf16_t* Vt  = (bf16_t*)alloc((size_t)H_ * 32 * T_ * 2);
  bf16_t* Ob  = (bf16_t*)alloc((size_t)T_ * AD_ * 2);

  auto cvt = [&](const float* s, bf16_t* d, int n) {
    k_cvt_f32_to_bf16<<<(n + 255) / 256, 256, 0, stream>>>(s, d, n);
  };
  cvt(x,   xb, T_ * DM_);
  cvt(qsw, Wb,                         128 * DM_);
  cvt(ksw, Wb + (size_t)128 * DM_,     128 * DM_);
  cvt(qgw, Wb + (size_t)256 * DM_,     256 * DM_);
  cvt(kgw, Wb + (size_t)512 * DM_,     256 * DM_);
  cvt(vw,  Wb + (size_t)768 * DM_,     512 * DM_);
  cvt(ow,  OWb, DM_ * AD_);

  k_gemm_bf16<<<(T_ / TM) * (QKVN_ / TN), 256, 0, stream>>>(xb, Wb, QKV, T_, QKVN_, DM_);
  k_pack_rope<<<(H_ * T_) / 256, 256, 0, stream>>>(QKV, lsc, poff, Qb, Kb, Vt);
  k_attn<<<(H_ * (T_ / 16)) / 8, 256, 0, stream>>>(Qb, Kb, Vt, Ob);
  k_gemm_bf16<<<(T_ / TM) * (DM_ / TN), 256, 0, stream>>>(Ob, OWb, out, T_, DM_, AD_);
}
